// ConvLayer_15161234555426
// MI455X (gfx1250) — matrix operand order repaired and checked by reading, not compiled, not run on hardware
//
#include <hip/hip_runtime.h>
#include <hip/hip_bf16.h>

#define N_NODES 50000
#define N_EDGES 800000
#define NODE_F  96
#define EDGE_F  64
#define D_IN    256
#define D_OUT   192
#define BN_EPS  1e-5f
#define STAT_SLOTS 32

typedef __bf16 bf16;
typedef __attribute__((ext_vector_type(16))) __bf16 v16bf;
typedef __attribute__((ext_vector_type(8)))  __bf16 v8bf;
typedef __attribute__((ext_vector_type(8)))  float  v8f;
typedef int v4i __attribute__((vector_size(16)));   // matches builtin param type
typedef __attribute__((address_space(1))) v4i* gptr_v4i;
typedef __attribute__((address_space(3))) v4i* lptr_v4i;

#if defined(__has_builtin)
#if __has_builtin(__builtin_amdgcn_global_load_async_to_lds_b128)
#define HAVE_ASYNC_LDS 1
#endif
#if __has_builtin(__builtin_amdgcn_s_wait_asynccnt)
#define HAVE_WAIT_ASYNC 1
#endif
#endif

// ---------------- prep kernels ----------------

__global__ void zero_f32(float* __restrict__ p, int n) {
  int i = blockIdx.x * blockDim.x + threadIdx.x;
  if (i < n) p[i] = 0.0f;
}

__global__ void cvt_node(const float* __restrict__ src, bf16* __restrict__ dst, int n) {
  int i = blockIdx.x * blockDim.x + threadIdx.x;
  if (i < n) dst[i] = (bf16)src[i];
}

// Pack W [256 x 192] f32 into per-lane WMMA B-fragment order:
// Wp[((kc*12+nt)*32 + lane)*16 + e] = W[(kc*32 + (e<8 ? e : e+8) + 8*(lane>>4)) * 192 + nt*16 + (lane&15)]
// (B 32x16 bf16 layout: lanes 0-15 hold K=0..7,16..23; lanes 16-31 hold K=8..15,24..31.)
__global__ void pack_w(const float* __restrict__ W, bf16* __restrict__ Wp) {
  int i = blockIdx.x * blockDim.x + threadIdx.x;
  if (i >= D_IN * D_OUT) return;
  int e    = i & 15;
  int lane = (i >> 4) & 31;
  int t    = i >> 9;            // kc*12 + nt
  int nt   = t % 12;
  int kc   = t / 12;
  int h    = lane >> 4;
  int n    = nt * 16 + (lane & 15);
  int K    = kc * 32 + (e < 8 ? e : e + 8) + 8 * h;
  Wp[i] = (bf16)W[K * D_OUT + n];
}

// ---------------- edge GEMM: y = concat(n1,n2,edge) @ W + b  (bf16 WMMA, f32 acc) ----------------

// A 16x32 bf16 fragment: lanes 0-15 and 16-31 both hold rows M=0..15;
// element e -> K = (e<8 ? e : e+8) + 8*h  => two contiguous 8-element runs.
static __device__ __forceinline__ v16bf load_a_frag(int kc,
                                                    const bf16* __restrict__ r1,
                                                    const bf16* __restrict__ r2,
                                                    const float* __restrict__ re,
                                                    int h) {
  v16bf a;
  if (kc < 6) {
    const bf16* r = (kc < 3) ? r1 : r2;
    int cb = ((kc < 3) ? kc : (kc - 3)) * 32 + h * 8;
    v8bf lo = *(const v8bf*)(r + cb);
    v8bf hi = *(const v8bf*)(r + cb + 16);
    a = __builtin_shufflevector(lo, hi, 0,1,2,3,4,5,6,7,8,9,10,11,12,13,14,15);
  } else {
    int cb = (kc - 6) * 32 + h * 8;
    float4 f0 = ((const float4*)(re + cb))[0];
    float4 f1 = ((const float4*)(re + cb))[1];
    float4 f2 = ((const float4*)(re + cb + 16))[0];
    float4 f3 = ((const float4*)(re + cb + 16))[1];
    a[0]=(bf16)f0.x; a[1]=(bf16)f0.y; a[2]=(bf16)f0.z; a[3]=(bf16)f0.w;
    a[4]=(bf16)f1.x; a[5]=(bf16)f1.y; a[6]=(bf16)f1.z; a[7]=(bf16)f1.w;
    a[8]=(bf16)f2.x; a[9]=(bf16)f2.y; a[10]=(bf16)f2.z; a[11]=(bf16)f2.w;
    a[12]=(bf16)f3.x; a[13]=(bf16)f3.y; a[14]=(bf16)f3.z; a[15]=(bf16)f3.w;
  }
  return a;
}

__launch_bounds__(256)
__global__ void gemm_edges(const bf16*  __restrict__ nodeb,
                           const float* __restrict__ edge,
                           const bf16*  __restrict__ Wp,
                           const float* __restrict__ bias,
                           const int*   __restrict__ idx1,
                           const int*   __restrict__ idx2,
                           bf16*        __restrict__ y,
                           float*       __restrict__ stat1) {
  __shared__ bf16  sW[D_IN * D_OUT];     // 96 KB: whole packed W, shared by 8 waves
  __shared__ float s_sum[D_OUT];
  __shared__ float s_sq[D_OUT];

  int tid = threadIdx.x;
  if (tid < D_OUT) { s_sum[tid] = 0.0f; s_sq[tid] = 0.0f; }

  // ---- stage packed W into LDS (6144 x 16B chunks, 24 per thread) ----
#if defined(HAVE_ASYNC_LDS)
  #pragma unroll
  for (int k = 0; k < 24; ++k) {
    int i = tid + k * 256;
    __builtin_amdgcn_global_load_async_to_lds_b128(
        (gptr_v4i)(v4i*)(const_cast<bf16*>(Wp) + i * 8),
        (lptr_v4i)(v4i*)(sW + i * 8),
        0, 0);
  }
#if defined(HAVE_WAIT_ASYNC)
  __builtin_amdgcn_s_wait_asynccnt(0);
#else
  asm volatile("s_wait_asynccnt 0x0" ::: "memory");
#endif
#else
  #pragma unroll
  for (int k = 0; k < 24; ++k) {
    int i = tid + k * 256;
    uint4 v = ((const uint4*)Wp)[i];
    ((uint4*)sW)[i] = v;
  }
#endif
  __syncthreads();

  int wave = tid >> 5;
  int lane = tid & 31;
  int h    = lane >> 4;        // half of wave
  int mr   = lane & 15;        // M row within 16
  int m0   = blockIdx.x * 128 + wave * 16;
  int m    = m0 + mr;          // this lane's A-matrix row (edge id)

  int i1 = idx1[m];
  int i2 = idx2[m];
  const bf16*  r1 = nodeb + (size_t)i1 * NODE_F;
  const bf16*  r2 = nodeb + (size_t)i2 * NODE_F;
  const float* re = edge  + (size_t)m  * EDGE_F;

  v8f acc[12] = {};

  // ---- software-pipelined K loop: A one k-chunk ahead, B one fragment ahead ----
  v16bf a_cur = load_a_frag(0, r1, r2, re, h);
  v16bf b_cur = *(const v16bf*)(sW + (lane << 4));   // kc=0, nt=0
  #pragma unroll
  for (int kc = 0; kc < 8; ++kc) {
    v16bf a_nxt = a_cur;
    if (kc < 7) a_nxt = load_a_frag(kc + 1, r1, r2, re, h);
    #pragma unroll
    for (int nt = 0; nt < 12; ++nt) {
      v16bf b_nxt = b_cur;
      if (nt < 11)
        b_nxt = *(const v16bf*)(sW + (((kc * 12 + nt + 1) * 32 + lane) << 4));
      else if (kc < 7)
        b_nxt = *(const v16bf*)(sW + ((((kc + 1) * 12) * 32 + lane) << 4));
      acc[nt] = __builtin_amdgcn_wmma_f32_16x16x32_bf16(
          false, a_cur, false, b_cur, (short)0, acc[nt], false, false);
      b_cur = b_nxt;
    }
    a_cur = a_nxt;
  }

  // ---- epilogue: bias, store y (bf16), per-column sum/sumsq for BN1 stats ----
  // C/D layout: col n = nt*16 + (lane&15); VGPR r holds row m0 + r + 8*h.
  #pragma unroll
  for (int nt = 0; nt < 12; ++nt) {
    int n = nt * 16 + mr;
    float bn = bias[n];
    float s = 0.0f, s2 = 0.0f;
    #pragma unroll
    for (int r = 0; r < 8; ++r) {
      float v = acc[nt][r] + bn;
      int mm = m0 + h * 8 + r;
      y[(size_t)mm * D_OUT + n] = (bf16)v;
      s += v; s2 += v * v;
    }
    s  += __shfl_xor(s, 16, 32);
    s2 += __shfl_xor(s2, 16, 32);
    if (h == 0) {
      atomicAdd(&s_sum[n], s);
      atomicAdd(&s_sq[n], s2);
    }
  }
  __syncthreads();
  if (tid < D_OUT) {
    int slot = blockIdx.x & (STAT_SLOTS - 1);
    atomicAdd(&stat1[slot * 2 * D_OUT + tid],         s_sum[tid]);
    atomicAdd(&stat1[slot * 2 * D_OUT + D_OUT + tid], s_sq[tid]);
  }
}

// ---------------- BN1 finalize: scale/shift per column ----------------

__global__ void bn1_finalize(const float* __restrict__ stat1,
                             const float* __restrict__ gamma1,
                             const float* __restrict__ beta1,
                             float* __restrict__ bn1) {
  int c = threadIdx.x;  // 192 threads
  float s = 0.0f, q = 0.0f;
  for (int k = 0; k < STAT_SLOTS; ++k) {
    s += stat1[k * 2 * D_OUT + c];
    q += stat1[k * 2 * D_OUT + D_OUT + c];
  }
  float mean = s / (float)N_EDGES;
  float var  = q / (float)N_EDGES - mean * mean;
  float rs   = rsqrtf(var + BN_EPS);
  float sc   = gamma1[c] * rs;
  bn1[c]         = sc;
  bn1[D_OUT + c] = beta1[c] - mean * sc;
}

// ---------------- message = sigmoid(gate)*softplus(conv); scatter-add ----------------

__global__ void msg_scatter(const bf16* __restrict__ y,
                            const float* __restrict__ bn1,
                            const int*  __restrict__ idx1,
                            float* __restrict__ sums,
                            float* __restrict__ counts) {
  int wave = (blockIdx.x * blockDim.x + threadIdx.x) >> 5;
  int lane = threadIdx.x & 31;
  int nwaves = (gridDim.x * blockDim.x) >> 5;
  for (int e = wave; e < N_EDGES; e += nwaves) {
    int en = e + nwaves;
    if (en < N_EDGES) {
      // prefetch next y row (streams at full lines); lowers to global_prefetch_b8
      __builtin_prefetch(y + (size_t)en * D_OUT + lane * 6, 0, 0);
    }
    int dst = idx1[e];
    const bf16* row = y + (size_t)e * D_OUT;
    #pragma unroll
    for (int j = 0; j < 3; ++j) {
      int c = lane + j * 32;
      float g = (float)row[c]          * bn1[c]          + bn1[D_OUT + c];
      float x = (float)row[NODE_F + c] * bn1[NODE_F + c] + bn1[D_OUT + NODE_F + c];
      float sig = 1.0f / (1.0f + __expf(-g));
      float sp  = fmaxf(x, 0.0f) + log1pf(__expf(-fabsf(x)));
      atomicAdd(&sums[(size_t)dst * NODE_F + c], sig * sp);
    }
    if (lane == 0) atomicAdd(&counts[dst], 1.0f);
  }
}

// ---------------- node mean (in-place) + BN2 stats ----------------

__global__ void node_mean_stats(float* __restrict__ sums,
                                const float* __restrict__ counts,
                                float* __restrict__ stat2) {
  __shared__ float s_sum[NODE_F];
  __shared__ float s_sq[NODE_F];
  int tid = threadIdx.x;
  if (tid < NODE_F) { s_sum[tid] = 0.0f; s_sq[tid] = 0.0f; }
  __syncthreads();
  int total = N_NODES * NODE_F;
  for (int i = blockIdx.x * blockDim.x + tid; i < total; i += gridDim.x * blockDim.x) {
    int n = i / NODE_F;
    int c = i - n * NODE_F;
    float cnt  = counts[n];
    float mean = sums[i] / fmaxf(cnt, 1.0f);
    sums[i] = mean;  // in-place: sums now holds per-node mean
    atomicAdd(&s_sum[c], mean);
    atomicAdd(&s_sq[c], mean * mean);
  }
  __syncthreads();
  if (tid < NODE_F) {
    int slot = blockIdx.x & (STAT_SLOTS - 1);
    atomicAdd(&stat2[slot * 2 * NODE_F + tid],          s_sum[tid]);
    atomicAdd(&stat2[slot * 2 * NODE_F + NODE_F + tid], s_sq[tid]);
  }
}

__global__ void bn2_finalize(const float* __restrict__ stat2,
                             const float* __restrict__ gamma2,
                             const float* __restrict__ beta2,
                             float* __restrict__ bn2) {
  int c = threadIdx.x;  // 96 threads
  float s = 0.0f, q = 0.0f;
  for (int k = 0; k < STAT_SLOTS; ++k) {
    s += stat2[k * 2 * NODE_F + c];
    q += stat2[k * 2 * NODE_F + NODE_F + c];
  }
  float mean = s / (float)N_NODES;
  float var  = q / (float)N_NODES - mean * mean;
  float rs   = rsqrtf(var + BN_EPS);
  float sc   = gamma2[c] * rs;
  bn2[c]          = sc;
  bn2[NODE_F + c] = beta2[c] - mean * sc;
}

// ---------------- final: softplus(node_fea + BN2(mean)) ----------------

__global__ void output_kernel(const float* __restrict__ node_fea,
                              const float* __restrict__ mean,  // == sums (in-place)
                              const float* __restrict__ bn2,
                              float* __restrict__ out) {
  int i = blockIdx.x * blockDim.x + threadIdx.x;
  if (i >= N_NODES * NODE_F) return;
  int c = i % NODE_F;
  float x = node_fea[i] + mean[i] * bn2[c] + bn2[NODE_F + c];
  out[i] = fmaxf(x, 0.0f) + log1pf(__expf(-fabsf(x)));
}

// ---------------- host ----------------

extern "C" void kernel_launch(void* const* d_in, const int* in_sizes, int n_in,
                              void* d_out, int out_size, void* d_ws, size_t ws_size,
                              hipStream_t stream) {
  const float* node_fea = (const float*)d_in[0];
  const float* edge_fea = (const float*)d_in[1];
  const float* W        = (const float*)d_in[2];
  const float* b        = (const float*)d_in[3];
  const float* gamma1   = (const float*)d_in[4];
  const float* beta1    = (const float*)d_in[5];
  const float* gamma2   = (const float*)d_in[6];
  const float* beta2    = (const float*)d_in[7];
  const int*   idx1     = (const int*)d_in[8];
  const int*   idx2     = (const int*)d_in[9];
  float* out = (float*)d_out;

  char* ws = (char*)d_ws;
  size_t off = 0;
  auto take = [&](size_t bytes) -> char* {
    char* p = ws + off;
    off += (bytes + 255) & ~(size_t)255;
    return p;
  };
  bf16*  nodeb  = (bf16*) take((size_t)N_NODES * NODE_F * sizeof(bf16));
  bf16*  Wp     = (bf16*) take((size_t)D_IN * D_OUT * sizeof(bf16));
  bf16*  y      = (bf16*) take((size_t)N_EDGES * D_OUT * sizeof(bf16));
  float* stat1  = (float*)take((size_t)STAT_SLOTS * 2 * D_OUT * sizeof(float));
  float* bn1    = (float*)take((size_t)2 * D_OUT * sizeof(float));
  float* sums   = (float*)take((size_t)N_NODES * NODE_F * sizeof(float));
  float* counts = (float*)take((size_t)N_NODES * sizeof(float));
  float* stat2  = (float*)take((size_t)STAT_SLOTS * 2 * NODE_F * sizeof(float));
  float* bn2    = (float*)take((size_t)2 * NODE_F * sizeof(float));

  // zero accumulators (deterministic per call; atomics accumulate into these)
  zero_f32<<<(N_NODES * NODE_F + 255) / 256, 256, 0, stream>>>(sums, N_NODES * NODE_F);
  zero_f32<<<(N_NODES + 255) / 256, 256, 0, stream>>>(counts, N_NODES);
  zero_f32<<<(STAT_SLOTS * 2 * D_OUT + 255) / 256, 256, 0, stream>>>(stat1, STAT_SLOTS * 2 * D_OUT);
  zero_f32<<<(STAT_SLOTS * 2 * NODE_F + 255) / 256, 256, 0, stream>>>(stat2, STAT_SLOTS * 2 * NODE_F);

  cvt_node<<<(N_NODES * NODE_F + 255) / 256, 256, 0, stream>>>(node_fea, nodeb, N_NODES * NODE_F);
  pack_w<<<(D_IN * D_OUT + 255) / 256, 256, 0, stream>>>(W, Wp);

  gemm_edges<<<N_EDGES / 128, 256, 0, stream>>>(nodeb, edge_fea, Wp, b, idx1, idx2, y, stat1);
  bn1_finalize<<<1, D_OUT, 0, stream>>>(stat1, gamma1, beta1, bn1);
  msg_scatter<<<2048, 256, 0, stream>>>(y, bn1, idx1, sums, counts);
  node_mean_stats<<<1024, 256, 0, stream>>>(sums, counts, stat2);
  bn2_finalize<<<1, NODE_F, 0, stream>>>(stat2, gamma2, beta2, bn2);
  output_kernel<<<(N_NODES * NODE_F + 255) / 256, 256, 0, stream>>>(node_fea, sums, bn2, out);
}
